// Vicon_25142738551252
// MI455X (gfx1250) — matrix operand order repaired and checked by reading, not compiled, not run on hardware
//
#include <hip/hip_runtime.h>
#include <hip/hip_bf16.h>

typedef __bf16 bf16;
typedef __attribute__((ext_vector_type(16))) __bf16 v16bf;
typedef __attribute__((ext_vector_type(8)))  float  v8f;

__device__ __forceinline__ bf16 to_bf16(float f) {
  unsigned u = __builtin_bit_cast(unsigned, f);
  unsigned r = u + 0x7FFFu + ((u >> 16) & 1u);
  return __builtin_bit_cast(bf16, (unsigned short)(r >> 16));
}

struct U8x { unsigned u[8]; };

// ---------------------------------------------------------------------------
// Generic bf16 WMMA GEMM:  C(MxN,f32) = A(MxK,bf16) * op(B) + bias + residual
//   transB==0:  B is KxN row-major (ldb)
//   transB==1:  B is NxK row-major (ldb)  (computes A * B^T)
// Block tile 128x64x32; 8 waves; each wave a 32x32 patch (2x2 WMMA 16x16x32).
// A tile staged via GLOBAL_LOAD_ASYNC_TO_LDS_B128 (ASYNCcnt); B tile staged
// through registers into a packed k-pair layout so fragments load as b32.
// All call sites have M%128==0, N%64==0, K%32==0 and 16B-aligned operands.
// ---------------------------------------------------------------------------
#define BM 128
#define BN 64
#define BK 32

__global__ __launch_bounds__(256) void gemm_bf16_wmma(
    const bf16* __restrict__ A, int lda,
    const bf16* __restrict__ B, int ldb, int transB,
    float* __restrict__ C, int ldc,
    const float* __restrict__ bias,
    const float* __restrict__ residual, int ldr,
    int M, int N, int K)
{
  // A tile: row-major bf16 => uint pairs.  16 data uints + 4 pad per row.
  __shared__ unsigned Au[BM][20];
  // B tile: packed k-pairs: Bs2[k/2][n] = {B[2k',n] , B[2k'+1,n]}
  __shared__ unsigned Bs2[BK / 2][BN + 2];

  const int tid  = threadIdx.x;
  const int lane = tid & 31;
  const int wave = tid >> 5;
  const int wm   = wave & 3;   // wave M position (0..3) * 32
  const int wn   = wave >> 2;  // wave N position (0..1) * 32
  const int half = lane >> 4;
  const int l16  = lane & 15;

  const int m0 = blockIdx.y * BM;
  const int n0 = blockIdx.x * BN;

  // ---- per-thread A-fill slots: two 16B chunks per k-step ----
  const int arow0 = tid >> 2;           // 0..63 (and +64)
  const int acu   = (tid & 3) * 4;      // uint column in Au (4 uints = 16B)
  const bf16* gA0 = A + (size_t)(m0 + arow0) * lda + acu * 2;
  const bf16* gA1 = A + (size_t)(m0 + arow0 + 64) * lda + acu * 2;
  const unsigned ldsA0 = (unsigned)(size_t)(void*)&Au[arow0][acu];
  const unsigned ldsA1 = (unsigned)(size_t)(void*)&Au[arow0 + 64][acu];

  // ---- per-thread B-fill slots ----
  // NN: (kpair, npair): kp in {tid>>5, tid>>5 + 8}, n = (tid&31)*2
  const int kpN = tid >> 5;
  const int npN = (tid & 31) * 2;
  const bf16* gB0 = B + (size_t)(2 * kpN) * ldb + n0 + npN;
  const bf16* gB1 = B + (size_t)(2 * (kpN + 8)) * ldb + n0 + npN;
  // NT: kp = tid&15, n = tid>>4 (+16 per s)
  const int kpT = tid & 15;
  const int nT  = tid >> 4;
  const bf16* gT = B + (size_t)(n0 + nT) * ldb + 2 * kpT;

  v8f acc[2][2];
#pragma unroll
  for (int i = 0; i < 2; ++i)
#pragma unroll
    for (int j = 0; j < 2; ++j)
#pragma unroll
      for (int e = 0; e < 8; ++e) acc[i][j][e] = 0.0f;

  for (int k0 = 0; k0 < K; k0 += BK) {
    // ---- A tile: async copy global -> LDS (512B/wave-inst, no VGPR data) ----
    asm volatile("global_load_async_to_lds_b128 %0, %1, off"
                 :: "v"(ldsA0), "v"(gA0) : "memory");
    asm volatile("global_load_async_to_lds_b128 %0, %1, off"
                 :: "v"(ldsA1), "v"(gA1) : "memory");
    gA0 += BK;
    gA1 += BK;

    // ---- B tile: register-staged swizzle into packed k-pair layout ----
    if (!transB) {
      const unsigned u00 = *(const unsigned*)gB0;
      const unsigned u01 = *(const unsigned*)(gB0 + ldb);
      const unsigned u10 = *(const unsigned*)gB1;
      const unsigned u11 = *(const unsigned*)(gB1 + ldb);
      gB0 += (size_t)BK * ldb;
      gB1 += (size_t)BK * ldb;
      Bs2[kpN][npN]         = (u00 & 0xffffu) | (u01 << 16);
      Bs2[kpN][npN + 1]     = (u00 >> 16) | (u01 & 0xffff0000u);
      Bs2[kpN + 8][npN]     = (u10 & 0xffffu) | (u11 << 16);
      Bs2[kpN + 8][npN + 1] = (u10 >> 16) | (u11 & 0xffff0000u);
    } else {
      unsigned t0 = *(const unsigned*)(gT);
      unsigned t1 = *(const unsigned*)(gT + (size_t)16 * ldb);
      unsigned t2 = *(const unsigned*)(gT + (size_t)32 * ldb);
      unsigned t3 = *(const unsigned*)(gT + (size_t)48 * ldb);
      gT += BK;
      Bs2[kpT][nT]      = t0;
      Bs2[kpT][nT + 16] = t1;
      Bs2[kpT][nT + 32] = t2;
      Bs2[kpT][nT + 48] = t3;
    }

    asm volatile("s_wait_asynccnt 0x0" ::: "memory");
    __syncthreads();

    // ---- fragments (CDNA5 16-bit operand layouts), loaded as uints ----
    v16bf afrag[2], bfrag[2];
#pragma unroll
    for (int sm = 0; sm < 2; ++sm) {
      U8x t;
      const int arow = wm * 32 + sm * 16 + l16;   // M = lane%16
#pragma unroll
      for (int v = 0; v < 8; ++v) {
        const int kidx = ((v < 4) ? 0 : 8) + half * 4 + (v & 3);
        t.u[v] = Au[arow][kidx];
      }
      afrag[sm] = __builtin_bit_cast(v16bf, t);
    }
#pragma unroll
    for (int sn = 0; sn < 2; ++sn) {
      U8x t;
      const int bcol = wn * 32 + sn * 16 + l16;   // N = lane%16
#pragma unroll
      for (int v = 0; v < 8; ++v) t.u[v] = Bs2[((v < 4) ? 0 : 8) + half * 4 + (v & 3)][bcol];
      bfrag[sn] = __builtin_bit_cast(v16bf, t);
    }

#pragma unroll
    for (int sm = 0; sm < 2; ++sm)
#pragma unroll
      for (int sn = 0; sn < 2; ++sn)
        acc[sm][sn] = __builtin_amdgcn_wmma_f32_16x16x32_bf16(
            false, afrag[sm], false, bfrag[sn], (short)0, acc[sm][sn],
            false, false);

    __syncthreads();
  }

  // ---- epilogue: C layout row = half*8 + r, col = lane%16 ----
#pragma unroll
  for (int sm = 0; sm < 2; ++sm)
#pragma unroll
    for (int sn = 0; sn < 2; ++sn) {
      const int col = n0 + wn * 32 + sn * 16 + l16;
#pragma unroll
      for (int r = 0; r < 8; ++r) {
        const int row = m0 + wm * 32 + sm * 16 + half * 8 + r;
        float val = acc[sm][sn][r];
        if (bias)     val += bias[col];
        if (residual) val += residual[(size_t)row * ldr + col];
        C[(size_t)row * ldc + col] = val;
      }
    }
}

// ---------------------------------------------------------------------------
// Elementwise / reduction kernels
// ---------------------------------------------------------------------------
__global__ void cvt_f32_bf16(const float* __restrict__ in, bf16* __restrict__ out, size_t n) {
  size_t i = (size_t)blockIdx.x * blockDim.x + threadIdx.x;
  size_t stride = (size_t)gridDim.x * blockDim.x;
  for (; i < n; i += stride) out[i] = to_bf16(in[i]);
}

__device__ __forceinline__ float block_sum(float v, float* sh) {
  int tid = threadIdx.x;
  sh[tid] = v; __syncthreads();
  for (int s = 128; s > 0; s >>= 1) { if (tid < s) sh[tid] += sh[tid + s]; __syncthreads(); }
  float r = sh[0]; __syncthreads(); return r;
}
__device__ __forceinline__ float block_max(float v, float* sh) {
  int tid = threadIdx.x;
  sh[tid] = v; __syncthreads();
  for (int s = 128; s > 0; s >>= 1) { if (tid < s) sh[tid] = fmaxf(sh[tid], sh[tid + s]); __syncthreads(); }
  float r = sh[0]; __syncthreads(); return r;
}

// one block (256 threads) per row; LayerNorm -> bf16
__global__ __launch_bounds__(256) void layernorm_bf16(
    const float* __restrict__ X, const float* __restrict__ s, const float* __restrict__ b,
    bf16* __restrict__ H, int D)
{
  __shared__ float sh[256];
  const int row = blockIdx.x;
  const float* x = X + (size_t)row * D;
  float acc = 0.f;
  for (int c = threadIdx.x; c < D; c += 256) acc += x[c];
  const float mean = block_sum(acc, sh) / (float)D;
  float vacc = 0.f;
  for (int c = threadIdx.x; c < D; c += 256) { float d = x[c] - mean; vacc += d * d; }
  const float var  = block_sum(vacc, sh) / (float)D;
  const float rstd = rsqrtf(var + 1e-5f);
  bf16* h = H + (size_t)row * D;
  for (int c = threadIdx.x; c < D; c += 256)
    h[c] = to_bf16((x[c] - mean) * rstd * s[c] + b[c]);
}

// one block per query row; scale + block-causal mask + softmax -> bf16 probs
__global__ __launch_bounds__(256) void softmax_mask_bf16(
    const float* __restrict__ S, bf16* __restrict__ P, int n, float scale)
{
  __shared__ float sh[256];
  const int q = blockIdx.x;
  const int kmax = ((q >> 8) + 1) << 8;   // allowed: kblock <= qblock (block = 256)
  const float* s = S + (size_t)q * n;
  bf16* p = P + (size_t)q * n;
  float mx = -1e30f;
  for (int c = threadIdx.x; c < kmax; c += 256) mx = fmaxf(mx, s[c] * scale);
  mx = block_max(mx, sh);
  float sum = 0.f;
  for (int c = threadIdx.x; c < kmax; c += 256) sum += __expf(s[c] * scale - mx);
  sum = block_sum(sum, sh);
  const float inv = 1.0f / sum;
  for (int c = threadIdx.x; c < n; c += 256)
    p[c] = (c < kmax) ? to_bf16(__expf(s[c] * scale - mx) * inv) : to_bf16(0.0f);
}

// tanh-approx GELU (jax.nn.gelu default), fp32 -> bf16
__global__ void gelu_bf16(const float* __restrict__ in, bf16* __restrict__ out, size_t n) {
  size_t i = (size_t)blockIdx.x * blockDim.x + threadIdx.x;
  size_t stride = (size_t)gridDim.x * blockDim.x;
  for (; i < n; i += stride) {
    float x = in[i];
    float t = tanhf(0.7978845608028654f * (x + 0.044715f * x * x * x));
    out[i] = to_bf16(0.5f * x * (1.0f + t));
  }
}

// feat += patch_pos[row%256] + func_pos[row/256]   (feat: 2560 x 1024)
__global__ void add_pos(float* __restrict__ x, const float* __restrict__ pp,
                        const float* __restrict__ fp) {
  size_t i = (size_t)blockIdx.x * 256 + threadIdx.x;
  if (i >= 2560ULL * 1024ULL) return;
  int row = (int)(i >> 10), col = (int)(i & 1023);
  x[i] += pp[(size_t)((row & 255) << 10) | col] + fp[(size_t)((row >> 8) << 10) | col];
}

// interleaved patchify of f,g -> P bf16 (2560 x 192)
__global__ void patchify_bf16(const float* __restrict__ f, const float* __restrict__ g,
                              bf16* __restrict__ P) {
  size_t i = (size_t)blockIdx.x * 256 + threadIdx.x;
  if (i >= 2560ULL * 192ULL) return;
  int row = (int)(i / 192), col = (int)(i % 192);
  int img = row >> 8, tok = row & 255;
  int pi = tok >> 4, pj = tok & 15;
  int ch = col >> 6, rem = col & 63;
  int iy = rem >> 3, ix = rem & 7;
  int pair = img >> 1, which = img & 1;
  const float* src = which ? g : f;
  P[i] = to_bf16(src[(((size_t)pair * 3 + ch) * 128 + (pi * 8 + iy)) * 128 + (pj * 8 + ix)]);
}

// gather condition-token rows (which==0) -> bf16 (1280 x 1024)
__global__ void gather_cond_bf16(const float* __restrict__ x, bf16* __restrict__ xc) {
  size_t i = (size_t)blockIdx.x * 256 + threadIdx.x;
  if (i >= 1280ULL * 1024ULL) return;
  int row = (int)(i >> 10), col = (int)(i & 1023);
  int pair = row >> 8, t = row & 255;
  xc[i] = to_bf16(x[((size_t)pair * 512 + t) * 1024 + col]);
}

// depatchify post_out (1280 x 192) -> out (5,3,128,128) flat
__global__ void depatchify_k(const float* __restrict__ po, float* __restrict__ out) {
  size_t i = (size_t)blockIdx.x * 256 + threadIdx.x;
  if (i >= 245760ULL) return;
  int xx = (int)(i & 127), y = (int)((i >> 7) & 127);
  int ch = (int)((i >> 14) % 3), pair = (int)(i / (3 * 128 * 128));
  int pi = y >> 3, iy = y & 7, pj = xx >> 3, ix = xx & 7;
  out[i] = po[((size_t)pair * 256 + pi * 16 + pj) * 192 + ch * 64 + iy * 8 + ix];
}

// ---------------------------------------------------------------------------
// Host orchestration
// ---------------------------------------------------------------------------
static inline void* ws_alloc(char* base, size_t& off, size_t bytes) {
  off = (off + 255) & ~(size_t)255;
  void* p = base + off;
  off += bytes;
  return p;
}

extern "C" void kernel_launch(void* const* d_in, const int* in_sizes, int n_in,
                              void* d_out, int out_size, void* d_ws, size_t ws_size,
                              hipStream_t stream) {
  (void)in_sizes; (void)n_in; (void)out_size; (void)ws_size;
  const int S = 2560, D = 1024, DFF = 4096, HEADS = 16, L = 4, PIN = 192, MC = 1280;
  const float scale = 0.125f;  // 64^-0.5

  const float* f_in      = (const float*)d_in[0];
  const float* g_in      = (const float*)d_in[1];
  const float* pre_W     = (const float*)d_in[2];
  const float* pre_b     = (const float*)d_in[3];
  const float* post_W    = (const float*)d_in[4];
  const float* post_b    = (const float*)d_in[5];
  const float* patch_pos = (const float*)d_in[6];
  const float* func_pos  = (const float*)d_in[7];
  const float* Wqkv      = (const float*)d_in[8];
  const float* bqkv      = (const float*)d_in[9];
  const float* Wo        = (const float*)d_in[10];
  const float* bo        = (const float*)d_in[11];
  const float* ln1_s     = (const float*)d_in[12];
  const float* ln1_b     = (const float*)d_in[13];
  const float* W1        = (const float*)d_in[14];
  const float* b1        = (const float*)d_in[15];
  const float* W2        = (const float*)d_in[16];
  const float* b2        = (const float*)d_in[17];
  const float* ln2_s     = (const float*)d_in[18];
  const float* ln2_b     = (const float*)d_in[19];

  char* ws = (char*)d_ws;
  size_t off = 0;
  bf16*  wqkv_b  = (bf16*) ws_alloc(ws, off, (size_t)L * D * 3 * D * 2);
  bf16*  wo_b    = (bf16*) ws_alloc(ws, off, (size_t)L * D * D * 2);
  bf16*  w1_b    = (bf16*) ws_alloc(ws, off, (size_t)L * D * DFF * 2);
  bf16*  w2_b    = (bf16*) ws_alloc(ws, off, (size_t)L * DFF * D * 2);
  bf16*  preW_b  = (bf16*) ws_alloc(ws, off, (size_t)PIN * D * 2);
  bf16*  postW_b = (bf16*) ws_alloc(ws, off, (size_t)D * PIN * 2);
  bf16*  P       = (bf16*) ws_alloc(ws, off, (size_t)S * PIN * 2);
  float* x       = (float*)ws_alloc(ws, off, (size_t)S * D * 4);
  bf16*  h       = (bf16*) ws_alloc(ws, off, (size_t)S * D * 2);
  float* qkv     = (float*)ws_alloc(ws, off, (size_t)S * 3 * D * 4);
  bf16*  qkvb    = (bf16*) ws_alloc(ws, off, (size_t)S * 3 * D * 2);
  float* scores  = (float*)ws_alloc(ws, off, (size_t)S * S * 4);
  bf16*  probs   = (bf16*) ws_alloc(ws, off, (size_t)S * S * 2);
  float* attno   = (float*)ws_alloc(ws, off, (size_t)S * D * 4);
  bf16*  attnob  = (bf16*) ws_alloc(ws, off, (size_t)S * D * 2);
  float* mlp     = (float*)ws_alloc(ws, off, (size_t)S * DFF * 4);
  bf16*  mlpb    = (bf16*) ws_alloc(ws, off, (size_t)S * DFF * 2);
  bf16*  xc      = (bf16*) ws_alloc(ws, off, (size_t)MC * D * 2);
  float* post_o  = (float*)ws_alloc(ws, off, (size_t)MC * PIN * 4);

  // ---- weight conversion fp32 -> bf16 ----
  cvt_f32_bf16<<<2048, 256, 0, stream>>>(Wqkv,   wqkv_b,  (size_t)L * D * 3 * D);
  cvt_f32_bf16<<<2048, 256, 0, stream>>>(Wo,     wo_b,    (size_t)L * D * D);
  cvt_f32_bf16<<<2048, 256, 0, stream>>>(W1,     w1_b,    (size_t)L * D * DFF);
  cvt_f32_bf16<<<2048, 256, 0, stream>>>(W2,     w2_b,    (size_t)L * DFF * D);
  cvt_f32_bf16<<<512,  256, 0, stream>>>(pre_W,  preW_b,  (size_t)PIN * D);
  cvt_f32_bf16<<<512,  256, 0, stream>>>(post_W, postW_b, (size_t)D * PIN);

  // ---- patchify + pre_proj + pos encodings ----
  patchify_bf16<<<(S * PIN + 255) / 256, 256, 0, stream>>>(f_in, g_in, P);
  gemm_bf16_wmma<<<dim3(D / BN, S / BM), 256, 0, stream>>>(
      P, PIN, preW_b, D, 0, x, D, pre_b, nullptr, 0, S, D, PIN);
  add_pos<<<(S * D + 255) / 256, 256, 0, stream>>>(x, patch_pos, func_pos);

  // ---- transformer layers ----
  for (int l = 0; l < L; ++l) {
    const bf16*  wqkv_l = wqkv_b + (size_t)l * D * 3 * D;
    const bf16*  wo_l   = wo_b   + (size_t)l * D * D;
    const bf16*  w1_l   = w1_b   + (size_t)l * D * DFF;
    const bf16*  w2_l   = w2_b   + (size_t)l * DFF * D;
    const float* bqkv_l = bqkv + (size_t)l * 3 * D;
    const float* bo_l   = bo   + (size_t)l * D;
    const float* b1_l   = b1   + (size_t)l * DFF;
    const float* b2_l   = b2   + (size_t)l * D;

    // LN1 -> h ; qkv = h @ Wqkv + bqkv
    layernorm_bf16<<<S, 256, 0, stream>>>(x, ln1_s + (size_t)l * D, ln1_b + (size_t)l * D, h, D);
    gemm_bf16_wmma<<<dim3(3 * D / BN, S / BM), 256, 0, stream>>>(
        h, D, wqkv_l, 3 * D, 0, qkv, 3 * D, bqkv_l, nullptr, 0, S, 3 * D, D);
    cvt_f32_bf16<<<2048, 256, 0, stream>>>(qkv, qkvb, (size_t)S * 3 * D);

    // attention per head: scores = Q K^T ; masked softmax ; o = P V
    for (int hh = 0; hh < HEADS; ++hh) {
      const bf16* Q = qkvb + hh * 64;
      const bf16* K = qkvb + D + hh * 64;
      const bf16* V = qkvb + 2 * D + hh * 64;
      gemm_bf16_wmma<<<dim3(S / BN, S / BM), 256, 0, stream>>>(
          Q, 3 * D, K, 3 * D, 1, scores, S, nullptr, nullptr, 0, S, S, 64);
      softmax_mask_bf16<<<S, 256, 0, stream>>>(scores, probs, S, scale);
      gemm_bf16_wmma<<<dim3(64 / BN, S / BM), 256, 0, stream>>>(
          probs, S, V, 3 * D, 0, attno + hh * 64, D, nullptr, nullptr, 0, S, 64, S);
    }
    cvt_f32_bf16<<<2048, 256, 0, stream>>>(attno, attnob, (size_t)S * D);

    // x = x + o @ Wo + bo   (residual folded into epilogue, in-place safe)
    gemm_bf16_wmma<<<dim3(D / BN, S / BM), 256, 0, stream>>>(
        attnob, D, wo_l, D, 0, x, D, bo_l, x, D, S, D, D);

    // LN2 -> h ; mlp = gelu(h @ W1 + b1) ; x = x + mlp @ W2 + b2
    layernorm_bf16<<<S, 256, 0, stream>>>(x, ln2_s + (size_t)l * D, ln2_b + (size_t)l * D, h, D);
    gemm_bf16_wmma<<<dim3(DFF / BN, S / BM), 256, 0, stream>>>(
        h, D, w1_l, DFF, 0, mlp, DFF, b1_l, nullptr, 0, S, DFF, D);
    gelu_bf16<<<2048, 256, 0, stream>>>(mlp, mlpb, (size_t)S * DFF);
    gemm_bf16_wmma<<<dim3(D / BN, S / BM), 256, 0, stream>>>(
        mlpb, DFF, w2_l, D, 0, x, D, b2_l, x, D, S, D, DFF);
  }

  // ---- post projection + depatchify ----
  gather_cond_bf16<<<(MC * D + 255) / 256, 256, 0, stream>>>(x, xc);
  gemm_bf16_wmma<<<dim3(PIN / BN, MC / BM), 256, 0, stream>>>(
      xc, D, postW_b, PIN, 0, post_o, PIN, post_b, nullptr, 0, MC, PIN, D);
  depatchify_k<<<245760 / 256, 256, 0, stream>>>(post_o, (float*)d_out);
}
